// PropNet_72619307041248
// MI455X (gfx1250) — compile-verified, matrix operand order checked by reading, not run on hardware
//
#include <hip/hip_runtime.h>

// ---------------------------------------------------------------------------
// PropNet on MI455X (gfx1250): bf16 WMMA GEMMs (32x32 per wave, LDS-staged
// weights via global_load_async_to_lds_b128) + index-based gather/scatter.
// ---------------------------------------------------------------------------

typedef __attribute__((ext_vector_type(16))) __bf16 v16bf;
typedef __attribute__((ext_vector_type(8)))  __bf16 v8bf;
typedef __attribute__((ext_vector_type(8)))  float  v8f;

#define B_   4
#define N_   1024
#define R_   16384
#define DINP 32    // 12  -> 32
#define EP   224   // 200 -> 224
#define RP3  320   // 300 -> 320
#define RELK 32    // 25  -> 32
#define RPK  768   // 300|200|200 concat -> 320+224+224
#define PPK  448   // 200|200 concat -> 224+224
#define PRN  32    // 3 -> 32 (N blocking is 32-wide)
#define LDSPAD 8   // bf16 elements of row padding in LDS (bank stagger)
#define KMAX 768

// ---------------------------------------------------------------------------
// GEMM: C = act(A[M,Kpad] @ W + bias), weights pre-transposed+padded to
// Wt[Npad][Kpad] (bf16). Block = 8 waves; the whole block shares one 32-row
// weight slab staged into LDS with async copies; each wave computes a 32x32
// output tile (4 x v_wmma_f32_16x16x32_bf16 accumulators).
//   grid.x = (Npad/32) * (M/256);  nt2 = blockIdx % Ntiles2
// A-fragment (16-bit 16x32): lane m = lane&15, half = lane>>4,
//   elems 0..7  = K[k + 8*half .. +7], elems 8..15 = K[k+16+8*half .. +7]
// C/D: VGPR j -> row m = mbase + j + 8*half, col n.
// ---------------------------------------------------------------------------
__global__ __launch_bounds__(256)
void k_gemm_bf16(const __bf16* __restrict__ A, int lda,
                 const __bf16* __restrict__ Wt,
                 const float* __restrict__ bias, int Nout,
                 __bf16* __restrict__ Cb, int ldcb,
                 float* __restrict__ Cf, int ldcf,
                 int Ntiles2, int Kpad, int relu)
{
    __shared__ __bf16 bs[32 * (KMAX + LDSPAD)];

    const int nt2  = blockIdx.x % Ntiles2;
    const int mg   = blockIdx.x / Ntiles2;
    const int wave = threadIdx.x >> 5;
    const int lane = threadIdx.x & 31;
    const int half = lane >> 4;
    const int lq   = lane & 15;
    const int ldk  = Kpad + LDSPAD;

    // -- cooperative async stage: weight rows [nt2*32 .. +31] x Kpad -> LDS --
    {
        const __bf16* gw = Wt + (long)nt2 * 32 * Kpad;  // contiguous slab
        const int rowChunks = Kpad >> 3;                // 16B chunks per row
        const int total = 32 * rowChunks;
        for (int c = threadIdx.x; c < total; c += 256) {
            int r  = c / rowChunks;
            int cc = c - r * rowChunks;
            unsigned lo = (unsigned)(size_t)&bs[r * ldk + cc * 8];
            const __bf16* gp = gw + (long)r * Kpad + (long)cc * 8;
            asm volatile("global_load_async_to_lds_b128 %0, %1, off"
                         :: "v"(lo), "v"(gp) : "memory");
        }
        asm volatile("s_wait_asynccnt 0x0" ::: "memory");
        __syncthreads();
    }

    const int  mt2   = mg * 8 + wave;        // 32-row tile index in M
    const long arow0 = (long)(mt2 * 32 + lq) * lda;
    const long arow1 = arow0 + (long)16 * lda;
    const int  koff  = 8 * half;
    const __bf16* b0p = &bs[lq * ldk];
    const __bf16* b1p = &bs[(lq + 16) * ldk];

    const int nb = nt2 * 32;
    const int n0 = nb + lq;
    const int n1 = nb + 16 + lq;
    float bv0 = (bias && n0 < Nout) ? bias[n0] : 0.0f;
    float bv1 = (bias && n1 < Nout) ? bias[n1] : 0.0f;
    v8f acc00, acc01, acc10, acc11;
#pragma unroll
    for (int i = 0; i < 8; ++i) {
        acc00[i] = bv0; acc01[i] = bv1; acc10[i] = bv0; acc11[i] = bv1;
    }

    for (int k = 0; k < Kpad; k += 32) {
        v8bf x0 = *(const v8bf*)(A + arow0 + k + koff);
        v8bf x1 = *(const v8bf*)(A + arow0 + k + 16 + koff);
        v8bf y0 = *(const v8bf*)(A + arow1 + k + koff);
        v8bf y1 = *(const v8bf*)(A + arow1 + k + 16 + koff);
        v8bf u0 = *(const v8bf*)(b0p + k + koff);
        v8bf u1 = *(const v8bf*)(b0p + k + 16 + koff);
        v8bf w0 = *(const v8bf*)(b1p + k + koff);
        v8bf w1 = *(const v8bf*)(b1p + k + 16 + koff);
        v16bf a0 = __builtin_shufflevector(x0, x1, 0,1,2,3,4,5,6,7,8,9,10,11,12,13,14,15);
        v16bf a1 = __builtin_shufflevector(y0, y1, 0,1,2,3,4,5,6,7,8,9,10,11,12,13,14,15);
        v16bf bb0 = __builtin_shufflevector(u0, u1, 0,1,2,3,4,5,6,7,8,9,10,11,12,13,14,15);
        v16bf bb1 = __builtin_shufflevector(w0, w1, 0,1,2,3,4,5,6,7,8,9,10,11,12,13,14,15);
        acc00 = __builtin_amdgcn_wmma_f32_16x16x32_bf16(false, a0, false, bb0, (short)0, acc00, false, false);
        acc01 = __builtin_amdgcn_wmma_f32_16x16x32_bf16(false, a0, false, bb1, (short)0, acc01, false, false);
        acc10 = __builtin_amdgcn_wmma_f32_16x16x32_bf16(false, a1, false, bb0, (short)0, acc10, false, false);
        acc11 = __builtin_amdgcn_wmma_f32_16x16x32_bf16(false, a1, false, bb1, (short)0, acc11, false, false);
    }

    const int mb0 = mt2 * 32;
    const int mb1 = mb0 + 16;
#pragma unroll
    for (int j = 0; j < 8; ++j) {
        const int m0 = mb0 + j + 8 * half;
        const int m1 = mb1 + j + 8 * half;
        float v00 = acc00[j], v01 = acc01[j], v10 = acc10[j], v11 = acc11[j];
        if (relu) {
            v00 = v00 < 0.0f ? 0.0f : v00;
            v01 = v01 < 0.0f ? 0.0f : v01;
            v10 = v10 < 0.0f ? 0.0f : v10;
            v11 = v11 < 0.0f ? 0.0f : v11;
        }
        if (Cb) {
            Cb[(long)m0 * ldcb + n0] = (__bf16)v00;
            Cb[(long)m0 * ldcb + n1] = (__bf16)v01;
            Cb[(long)m1 * ldcb + n0] = (__bf16)v10;
            Cb[(long)m1 * ldcb + n1] = (__bf16)v11;
        }
        if (Cf) {
            if (n0 < Nout) {
                Cf[(long)m0 * ldcf + n0] = v00;
                Cf[(long)m1 * ldcf + n0] = v10;
            }
            if (n1 < Nout) {
                Cf[(long)m0 * ldcf + n1] = v01;
                Cf[(long)m1 * ldcf + n1] = v11;
            }
        }
    }
}

// ---------------------------------------------------------------------------
// Helpers
// ---------------------------------------------------------------------------
__global__ void k_zero32(unsigned* __restrict__ p, long n)
{
    long i = (long)blockIdx.x * blockDim.x + threadIdx.x;
    long stride = (long)gridDim.x * blockDim.x;
    for (; i < n; i += stride) p[i] = 0u;
}

// Recover recv/send indices from the dense one-hot Rr/Rs [B,N,R].
__global__ void k_extract_idx(const float* __restrict__ Rr,
                              const float* __restrict__ Rs,
                              int* __restrict__ recv, int* __restrict__ send,
                              long total)
{
    long i = (long)blockIdx.x * blockDim.x + threadIdx.x;
    long stride = (long)gridDim.x * blockDim.x;
    for (; i < total; i += stride) {
        float vr = Rr[i];
        float vs = Rs[i];
        if (vr != 0.0f || vs != 0.0f) {
            int  r = (int)(i % R_);
            long t = i / R_;
            int  n = (int)(t % N_);
            int  b = (int)(t / N_);
            if (vr != 0.0f) recv[(long)b * R_ + r] = n;
            if (vs != 0.0f) send[(long)b * R_ + r] = n;
        }
    }
}

// x = concat(attr,state,action) -> bf16 [B*N, 32] (cols 12..31 zero)
__global__ void k_build_x(const float* __restrict__ attr,
                          const float* __restrict__ state,
                          const float* __restrict__ action,
                          __bf16* __restrict__ xb)
{
    int i = blockIdx.x * blockDim.x + threadIdx.x;
    if (i >= B_ * N_) return;
    __bf16* row = xb + (long)i * DINP;
#pragma unroll
    for (int c = 0; c < 3; ++c) row[c]     = (__bf16)attr[(long)i * 3 + c];
#pragma unroll
    for (int c = 0; c < 6; ++c) row[3 + c] = (__bf16)state[(long)i * 6 + c];
#pragma unroll
    for (int c = 0; c < 3; ++c) row[9 + c] = (__bf16)action[(long)i * 3 + c];
#pragma unroll
    for (int c = 12; c < DINP; ++c) row[c] = (__bf16)0.0f;
}

// Wt[n*Kpad + kDst + k] = bf16(W[(rowStart+k)*ldW + n]) for k<rowCnt, n<Nc
__global__ void k_transpose_w(const float* __restrict__ W, int Nc, int ldW,
                              int rowStart, int rowCnt,
                              __bf16* __restrict__ Wt, int Kpad, int kDst)
{
    long total = (long)rowCnt * Nc;
    long i = (long)blockIdx.x * blockDim.x + threadIdx.x;
    long stride = (long)gridDim.x * blockDim.x;
    for (; i < total; i += stride) {
        int n = (int)(i % Nc);
        int k = (int)(i / Nc);
        Wt[(long)n * Kpad + kDst + k] = (__bf16)W[(long)(rowStart + k) * ldW + n];
    }
}

// rel_in[m, 0:12]=x[recv], [12:24]=x[send], [24]=Ra, [25:32]=0
__global__ void k_build_rel_in(const __bf16* __restrict__ xb,
                               const float* __restrict__ Ra,
                               const int* __restrict__ recv,
                               const int* __restrict__ send,
                               __bf16* __restrict__ relin)
{
    int m = blockIdx.x * blockDim.x + threadIdx.x;
    if (m >= B_ * R_) return;
    int b = m / R_;
    const __bf16* xr = xb + ((long)b * N_ + recv[m]) * DINP;
    const __bf16* xs = xb + ((long)b * N_ + send[m]) * DINP;
    __bf16* row = relin + (long)m * RELK;
#pragma unroll
    for (int c = 0; c < 12; ++c) { row[c] = xr[c]; row[12 + c] = xs[c]; }
    row[24] = (__bf16)Ra[m];
#pragma unroll
    for (int c = 25; c < RELK; ++c) row[c] = (__bf16)0.0f;
}

// rp_in[m, 320:544] = pe[recv[m]], rp_in[m, 544:768] = pe[send[m]]
__global__ void k_gather_eff(const __bf16* __restrict__ pe,
                             const int* __restrict__ recv,
                             const int* __restrict__ send,
                             __bf16* __restrict__ rpin)
{
    int m = blockIdx.x;
    int b = m / R_;
    const __bf16* pr = pe + ((long)b * N_ + recv[m]) * EP;
    const __bf16* ps = pe + ((long)b * N_ + send[m]) * EP;
    __bf16* row = rpin + (long)m * RPK;
    for (int c = threadIdx.x; c < EP; c += blockDim.x) {
        row[320 + c] = pr[c];
        row[544 + c] = ps[c];
    }
}

// effect_agg[b, recv[m], :] += rel_eff[m, :]  (f32 atomics)
__global__ void k_scatter_add(const __bf16* __restrict__ releff,
                              const int* __restrict__ recv,
                              float* __restrict__ agg)
{
    int m = blockIdx.x;
    int b = m / R_;
    const __bf16* row = releff + (long)m * EP;
    float* dst = agg + ((long)b * N_ + recv[m]) * EP;
    for (int c = threadIdx.x; c < 200; c += blockDim.x)
        atomicAdd(dst + c, (float)row[c]);
}

// pp_in[:, 224:448] = bf16(effect_agg)
__global__ void k_cvt_agg(const float* __restrict__ agg, __bf16* __restrict__ ppin)
{
    long total = (long)B_ * N_ * EP;
    long i = (long)blockIdx.x * blockDim.x + threadIdx.x;
    if (i >= total) return;
    long r = i / EP, c = i % EP;
    ppin[r * PPK + EP + c] = (__bf16)agg[i];
}

// ---------------------------------------------------------------------------
// Host side
// ---------------------------------------------------------------------------
static inline void* carve(char*& p, size_t bytes)
{
    void* r = (void*)p;
    p += (bytes + 255) & ~(size_t)255;
    return r;
}

static inline void gemm(hipStream_t s, const __bf16* A, int lda, const __bf16* Wt,
                        int Kpad, const float* bias, int Nout,
                        __bf16* Cb, int ldcb, float* Cf, int ldcf,
                        int M, int Npad, int relu)
{
    int Ntiles2 = Npad / 32;          // 32-wide N blocking
    int blocks  = Ntiles2 * (M / 256); // 8 waves x 32 rows per block
    k_gemm_bf16<<<blocks, 256, 0, s>>>(A, lda, Wt, bias, Nout, Cb, ldcb, Cf, ldcf,
                                       Ntiles2, Kpad, relu);
}

extern "C" void kernel_launch(void* const* d_in, const int* in_sizes, int n_in,
                              void* d_out, int out_size, void* d_ws, size_t ws_size,
                              hipStream_t stream)
{
    (void)in_sizes; (void)n_in; (void)out_size; (void)ws_size;

    const float* attr   = (const float*)d_in[0];
    const float* state  = (const float*)d_in[1];
    const float* action = (const float*)d_in[2];
    const float* Rr     = (const float*)d_in[3];
    const float* Rs     = (const float*)d_in[4];
    const float* Ra     = (const float*)d_in[5];
    const float* pe_w0  = (const float*)d_in[6];
    const float* pe_b0  = (const float*)d_in[7];
    const float* pe_w1  = (const float*)d_in[8];
    const float* pe_b1  = (const float*)d_in[9];
    const float* re_w0  = (const float*)d_in[10];
    const float* re_b0  = (const float*)d_in[11];
    const float* re_w1  = (const float*)d_in[12];
    const float* re_b1  = (const float*)d_in[13];
    const float* re_w2  = (const float*)d_in[14];
    const float* re_b2  = (const float*)d_in[15];
    const float* rp_w   = (const float*)d_in[16];
    const float* rp_b   = (const float*)d_in[17];
    const float* pp_w   = (const float*)d_in[18];
    const float* pp_b   = (const float*)d_in[19];
    const float* pr_w0  = (const float*)d_in[20];
    const float* pr_b0  = (const float*)d_in[21];
    const float* pr_w1  = (const float*)d_in[22];
    const float* pr_b1  = (const float*)d_in[23];
    float* out = (float*)d_out;               // [B,N,3] f32
    const int pstep = 2;                      // reference setup constant

    char* p = (char*)d_ws;

    // -- weight buffers (contiguous so one zero-fill covers all padding) ----
    char* wbase = p;
    __bf16* pe_w0t = (__bf16*)carve(p, (size_t)EP  * DINP * 2);
    __bf16* pe_w1t = (__bf16*)carve(p, (size_t)EP  * EP   * 2);
    __bf16* re_w0t = (__bf16*)carve(p, (size_t)RP3 * RELK * 2);
    __bf16* re_w1t = (__bf16*)carve(p, (size_t)RP3 * RP3  * 2);
    __bf16* re_w2t = (__bf16*)carve(p, (size_t)RP3 * RP3  * 2);
    __bf16* rp_wt  = (__bf16*)carve(p, (size_t)EP  * RPK  * 2);
    __bf16* pp_wt  = (__bf16*)carve(p, (size_t)EP  * PPK  * 2);
    __bf16* pr_w0t = (__bf16*)carve(p, (size_t)EP  * EP   * 2);
    __bf16* pr_w1t = (__bf16*)carve(p, (size_t)PRN * EP   * 2);
    size_t wbytes = (size_t)(p - wbase);

    // -- activations & scratch ----------------------------------------------
    int*    recv    = (int*)   carve(p, (size_t)B_ * R_ * 4);
    int*    send    = (int*)   carve(p, (size_t)B_ * R_ * 4);
    __bf16* xb      = (__bf16*)carve(p, (size_t)B_ * N_ * DINP * 2);
    __bf16* peh     = (__bf16*)carve(p, (size_t)B_ * N_ * EP   * 2);
    __bf16* rel_in  = (__bf16*)carve(p, (size_t)B_ * R_ * RELK * 2);
    __bf16* rh1     = (__bf16*)carve(p, (size_t)B_ * R_ * RP3  * 2);
    __bf16* rh2     = (__bf16*)carve(p, (size_t)B_ * R_ * RP3  * 2);
    __bf16* rp_in   = (__bf16*)carve(p, (size_t)B_ * R_ * RPK  * 2);
    __bf16* rel_eff = (__bf16*)carve(p, (size_t)B_ * R_ * EP   * 2);
    float*  eff_agg = (float*) carve(p, (size_t)B_ * N_ * EP   * 4);
    __bf16* pp_in   = (__bf16*)carve(p, (size_t)B_ * N_ * PPK  * 2);
    __bf16* pe_buf  = (__bf16*)carve(p, (size_t)B_ * N_ * EP   * 2);
    __bf16* pred_h  = (__bf16*)carve(p, (size_t)B_ * N_ * EP   * 2);

    // -- zero the padded weight region and initial pe -----------------------
    {
        long n = (long)(wbytes / 4);
        k_zero32<<<(int)((n + 255) / 256), 256, 0, stream>>>((unsigned*)wbase, n);
        long m = (long)B_ * N_ * EP / 2; // bf16 -> u32 count
        k_zero32<<<(int)((m + 255) / 256), 256, 0, stream>>>((unsigned*)pe_buf, m);
    }

    // -- transpose + pad weights into bf16 [Npad, Kpad] ----------------------
    #define TR(W, Nc, ldW, rs, rc, Wt, Kp, kd)                                   \
        k_transpose_w<<<(int)(((long)(rc)*(Nc) + 255) / 256), 256, 0, stream>>>( \
            W, Nc, ldW, rs, rc, Wt, Kp, kd)
    TR(pe_w0, 200, 200, 0,  12, pe_w0t, DINP, 0);
    TR(pe_w1, 200, 200, 0, 200, pe_w1t, EP,   0);
    TR(re_w0, 300, 300, 0,  25, re_w0t, RELK, 0);
    TR(re_w1, 300, 300, 0, 300, re_w1t, RP3,  0);
    TR(re_w2, 300, 300, 0, 300, re_w2t, RP3,  0);
    TR(rp_w,  200, 200, 0,   300, rp_wt, RPK, 0);
    TR(rp_w,  200, 200, 300, 200, rp_wt, RPK, 320);
    TR(rp_w,  200, 200, 500, 200, rp_wt, RPK, 544);
    TR(pp_w,  200, 200, 0,   200, pp_wt, PPK, 0);
    TR(pp_w,  200, 200, 200, 200, pp_wt, PPK, 224);
    TR(pr_w0, 200, 200, 0, 200, pr_w0t, EP, 0);
    TR(pr_w1,   3,   3, 0, 200, pr_w1t, EP, 0);
    #undef TR

    // -- recover indices from one-hot Rr/Rs ----------------------------------
    k_extract_idx<<<65536, 256, 0, stream>>>(Rr, Rs, recv, send,
                                             (long)B_ * N_ * R_);

    // -- build inputs --------------------------------------------------------
    k_build_x<<<(B_ * N_ + 255) / 256, 256, 0, stream>>>(attr, state, action, xb);
    k_build_rel_in<<<(B_ * R_ + 255) / 256, 256, 0, stream>>>(xb, Ra, recv, send, rel_in);

    const int MP = B_ * N_;   // 4096
    const int MR = B_ * R_;   // 65536

    // -- particle encoder: 12->200->200 (ReLU), second layer -> pp_in[:,0:224]
    gemm(stream, xb,  DINP, pe_w0t, DINP, pe_b0, 200, peh,   EP,  nullptr, 0, MP, EP, 1);
    gemm(stream, peh, EP,   pe_w1t, EP,   pe_b1, 200, pp_in, PPK, nullptr, 0, MP, EP, 1);

    // -- relation encoder: 25->300->300->300 (ReLU), last -> rp_in[:,0:320]
    gemm(stream, rel_in, RELK, re_w0t, RELK, re_b0, 300, rh1, RP3, nullptr, 0, MR, RP3, 1);
    gemm(stream, rh1,    RP3,  re_w1t, RP3,  re_b1, 300, rh2, RP3, nullptr, 0, MR, RP3, 1);
    gemm(stream, rh2,    RP3,  re_w2t, RP3,  re_b2, 300, rp_in, RPK, nullptr, 0, MR, RP3, 1);

    // -- propagation steps ---------------------------------------------------
    for (int t = 0; t < pstep; ++t) {
        long na = (long)B_ * N_ * EP;
        k_zero32<<<(int)((na + 255) / 256), 256, 0, stream>>>((unsigned*)eff_agg, na);
        k_gather_eff<<<MR, 256, 0, stream>>>(pe_buf, recv, send, rp_in);
        gemm(stream, rp_in, RPK, rp_wt, RPK, rp_b, 200, rel_eff, EP, nullptr, 0, MR, EP, 1);
        k_scatter_add<<<MR, 256, 0, stream>>>(rel_eff, recv, eff_agg);
        long nc = (long)B_ * N_ * EP;
        k_cvt_agg<<<(int)((nc + 255) / 256), 256, 0, stream>>>(eff_agg, pp_in);
        gemm(stream, pp_in, PPK, pp_wt, PPK, pp_b, 200, pe_buf, EP, nullptr, 0, MP, EP, 1);
    }

    // -- predictor: 200->200 (ReLU) -> 3, f32 out ----------------------------
    gemm(stream, pe_buf, EP, pr_w0t, EP, pr_b0, 200, pred_h, EP, nullptr, 0, MP, EP, 1);
    gemm(stream, pred_h, EP, pr_w1t, EP, pr_b1, 3,   nullptr, 0, out, 3,   MP, PRN, 0);
}
